// MultiHeadDistanceLayer_19146964205790
// MI455X (gfx1250) — matrix operand order repaired and checked by reading, not compiled, not run on hardware
//
#include <hip/hip_runtime.h>
#include <hip/hip_bf16.h>

typedef __attribute__((ext_vector_type(16))) _Float16 v16h;
typedef __attribute__((ext_vector_type(8)))  float    v8f;

#define BATCH  2
#define FEAT   512
#define LEN    2048
#define NH     8
#define HD     64
#define DMODEL 512  // NH*HD

// ---------------------------------------------------------------------------
// WMMA fragment helpers (layouts per CDNA5 ISA 7.12.2, wave32)
// A 16x32 f16: lane = m + 16*g (m = row). VGPR p (pair i=2p+e) holds
//   K = 16*(p/4) + 8*g + 2*(p%4) + e   -> contiguous K-pairs.
// B 32x16 f16: lane = c + 16*g (c = col). VGPR p holds K = 16*(p/4) + 8*g + 2*(p%4) + e, as for A.
// C/D 16x16 f32: element r in lane (n,g) is D[8*g + r][n].
// ---------------------------------------------------------------------------
__device__ __forceinline__ int a_koff(int p, int g) {
    return ((p >> 2) << 4) + (g << 3) + ((p & 3) << 1);
}

__device__ __forceinline__ v16h load_frag_a(const _Float16* rowp, int g) {
    v16h a;
#pragma unroll
    for (int p = 0; p < 8; ++p) {
        int o = a_koff(p, g);
        a[2 * p]     = rowp[o];
        a[2 * p + 1] = rowp[o + 1];
    }
    return a;
}

__device__ __forceinline__ v16h load_frag_b(const _Float16* colp, int g) {
    v16h b;
#pragma unroll
    for (int p = 0; p < 8; ++p) {
        int o = a_koff(p, g);
        b[2 * p]     = colp[o];
        b[2 * p + 1] = colp[o + 1];
    }
    return b;
}

__device__ __forceinline__ v8f wmma16(v16h a, v16h b, v8f c) {
    return __builtin_amdgcn_wmma_f32_16x16x32_f16(false, a, false, b,
                                                  (short)0, c, false, false);
}

// ---------------------------------------------------------------------------
// Kernel 1: qk_in = transpose(inputs) + positional encoding  (f32 -> f16)
//           plus f16 copies of Wq, Wk.
// ---------------------------------------------------------------------------
__global__ void k_prepare(const float* __restrict__ inp,
                          const float* __restrict__ Wq,
                          const float* __restrict__ Wk,
                          _Float16* __restrict__ qk_h,
                          _Float16* __restrict__ wq_h,
                          _Float16* __restrict__ wk_h) {
    int tid = blockIdx.x * blockDim.x + threadIdx.x;
    const int total = BATCH * LEN * FEAT;
    if (tid < total) {
        int f = tid % FEAT;
        int l = (tid / FEAT) % LEN;
        int b = tid / (FEAT * LEN);
        int j = f >> 1;  // dim_t uses floor(f/2)
        // pos = l * TEMPERATURE^(-2j/F);  ln(10000) = 9.210340371976184
        float ang = (float)l *
                    __expf(-(2.0f * (float)j / (float)FEAT) * 9.210340371976184f);
        float pe = (f & 1) ? __cosf(ang) : __sinf(ang);
        float x  = inp[(b * FEAT + f) * LEN + l];
        qk_h[tid] = (_Float16)(x + pe);
    }
    const int wtotal = DMODEL * FEAT;
    if (tid < wtotal) {
        wq_h[tid] = (_Float16)Wq[tid];
    } else if (tid < 2 * wtotal) {
        wk_h[tid - wtotal] = (_Float16)Wk[tid - wtotal];
    }
}

// ---------------------------------------------------------------------------
// Kernel 2: Q/K projection GEMM via WMMA.
//   q = qk_in @ Wq^T + bq ; k = qk_in @ Wk^T + bk
//   Q stored (B,H,L,64) f16; K stored both (B,H,L,64) and transposed (B,H,64,L).
//   One wave per 16x16 output tile; K-loop of 16 WMMAs over FEAT=512.
// ---------------------------------------------------------------------------
__global__ void k_proj(const _Float16* __restrict__ qk_h,
                       const _Float16* __restrict__ wq_h,
                       const _Float16* __restrict__ wk_h,
                       const float* __restrict__ bq,
                       const float* __restrict__ bk,
                       _Float16* __restrict__ q_h,
                       _Float16* __restrict__ kd_h,
                       _Float16* __restrict__ kt_h) {
    int wid  = blockIdx.x * (blockDim.x >> 5) + (threadIdx.x >> 5);
    int lane = threadIdx.x & 31;
    int g    = lane >> 4;
    int lc   = lane & 15;  // A-row within tile / B-col within tile

    const int TN = DMODEL / 16;       // 32
    const int TM = BATCH * LEN / 16;  // 256
    int tn = wid % TN;  wid /= TN;
    int tm = wid % TM;
    int qk = wid / TM;  // 0 = Q, 1 = K

    const _Float16* W    = qk ? wk_h : wq_h;
    const float*    bias = qk ? bk : bq;

    const _Float16* arow = qk_h + (size_t)(tm * 16 + lc) * FEAT;
    const _Float16* bcol = W    + (size_t)(tn * 16 + lc) * FEAT;

    v8f acc = {};
#pragma unroll
    for (int kk = 0; kk < FEAT; kk += 32) {
        v16h a = load_frag_a(arow + kk, g);
        v16h b = load_frag_b(bcol + kk, g);
        acc = wmma16(a, b, acc);
    }

    int n  = tn * 16 + lc;
    int h  = n >> 6;
    int dc = n & 63;
    float bv = bias[n];
#pragma unroll
    for (int r = 0; r < 8; ++r) {
        int row = tm * 16 + 8 * g + r;  // global row over B*L
        int b_  = row >> 11;
        int l   = row & (LEN - 1);
        _Float16 val = (_Float16)(acc[r] + bv);
        size_t bh = (size_t)(b_ * NH + h);
        if (qk == 0) {
            q_h[(bh * LEN + l) * HD + dc] = val;
        } else {
            kd_h[(bh * LEN + l) * HD + dc] = val;
            kt_h[(bh * HD + dc) * LEN + l] = val;
        }
    }
}

// ---------------------------------------------------------------------------
// Kernel 3: attention. One block (8 waves) per (b,h,q-tile of 16).
//   Phase 1: scores = Q K^T * d^-0.5 -- wave w owns key tiles w,w+8,...
//            (16 tiles => 16 v8f accumulators resident in VGPRs)
//   Phase 2: softmax (shfl intra-wave + LDS cross-wave reduction)
//   Phase 3: attn_out streamed with nontemporal stores; f16 probs -> LDS
//   Phase 4: dist = attn @ K via WMMA from LDS A-frags + Kt B-frags,
//            cross-wave f32 reduction in LDS.
// ---------------------------------------------------------------------------
__global__ void k_attn(const _Float16* __restrict__ q_h,
                       const _Float16* __restrict__ kd_h,
                       const _Float16* __restrict__ kt_h,
                       float* __restrict__ out_dist,
                       float* __restrict__ out_attn) {
    extern __shared__ char smem[];
    _Float16* attn_lds = (_Float16*)smem;                 // 16*2048*2   = 64 KiB
    float* distp  = (float*)(smem + 16 * LEN * 2);        // 8*16*64*4   = 32 KiB
    float* redmax = distp + 8 * 16 * 64;                  // 8*16*4
    float* redsum = redmax + 8 * 16;                      // 8*16*4

    int w    = threadIdx.x >> 5;
    int lane = threadIdx.x & 31;
    int g    = lane >> 4;
    int lc   = lane & 15;

    int qt = blockIdx.x & (LEN / 16 - 1);  // 0..127
    int bh = blockIdx.x >> 7;              // 0..15
    int b  = bh >> 3;
    int h  = bh & 7;
    int q0 = qt * 16;

    const _Float16* qbase  = q_h  + ((size_t)bh * LEN + q0) * HD;
    const _Float16* kdbase = kd_h + (size_t)bh * LEN * HD;
    const _Float16* ktbase = kt_h + (size_t)bh * HD * LEN;

    // Q tile A-fragments for d-chunks [0,32) and [32,64)
    v16h aQ0 = load_frag_a(qbase + (size_t)lc * HD + 0,  g);
    v16h aQ1 = load_frag_a(qbase + (size_t)lc * HD + 32, g);

    // ---- Phase 1: scores -------------------------------------------------
    v8f sc[16];
#pragma unroll
    for (int j = 0; j < 16; ++j) {
        int keyb = (w + 8 * j) * 16;
        const _Float16* kcol = kdbase + (size_t)(keyb + lc) * HD;
        v8f acc = {};
        acc = wmma16(aQ0, load_frag_b(kcol + 0,  g), acc);
        acc = wmma16(aQ1, load_frag_b(kcol + 32, g), acc);
#pragma unroll
        for (int r = 0; r < 8; ++r) acc[r] *= 0.125f;  // 1/sqrt(64)
        sc[j] = acc;
    }

    // ---- Phase 2: softmax ------------------------------------------------
    v8f vm = sc[0];
#pragma unroll
    for (int j = 1; j < 16; ++j)
#pragma unroll
        for (int r = 0; r < 8; ++r) vm[r] = fmaxf(vm[r], sc[j][r]);
#pragma unroll
    for (int mask = 1; mask <= 8; mask <<= 1)
#pragma unroll
        for (int r = 0; r < 8; ++r)
            vm[r] = fmaxf(vm[r], __shfl_xor(vm[r], mask, 32));
    if (lc == 0) {
#pragma unroll
        for (int r = 0; r < 8; ++r) redmax[w * 16 + 8 * g + r] = vm[r];
    }
    __syncthreads();
    float rowmax[8];
#pragma unroll
    for (int r = 0; r < 8; ++r) {
        float m = -3.0e38f;
        for (int w2 = 0; w2 < 8; ++w2)
            m = fmaxf(m, redmax[w2 * 16 + 8 * g + r]);
        rowmax[r] = m;
    }

    v8f vs = {};
#pragma unroll
    for (int j = 0; j < 16; ++j)
#pragma unroll
        for (int r = 0; r < 8; ++r) {
            float e = __expf(sc[j][r] - rowmax[r]);
            sc[j][r] = e;
            vs[r] += e;
        }
#pragma unroll
    for (int mask = 1; mask <= 8; mask <<= 1)
#pragma unroll
        for (int r = 0; r < 8; ++r) vs[r] += __shfl_xor(vs[r], mask, 32);
    if (lc == 0) {
#pragma unroll
        for (int r = 0; r < 8; ++r) redsum[w * 16 + 8 * g + r] = vs[r];
    }
    __syncthreads();
    float inv[8];
#pragma unroll
    for (int r = 0; r < 8; ++r) {
        float s = 0.0f;
        for (int w2 = 0; w2 < 8; ++w2) s += redsum[w2 * 16 + 8 * g + r];
        inv[r] = 1.0f / s;
    }

    // ---- Phase 3: attn_out (B,Lq,Lk,H) + f16 staging for dist ------------
#pragma unroll
    for (int j = 0; j < 16; ++j) {
        int k = (w + 8 * j) * 16 + lc;
#pragma unroll
        for (int r = 0; r < 8; ++r) {
            float p = sc[j][r] * inv[r];
            int q = q0 + 8 * g + r;
            __builtin_nontemporal_store(
                p, &out_attn[(((size_t)b * LEN + q) * LEN + k) * NH + h]);
            attn_lds[(8 * g + r) * LEN + k] = (_Float16)p;
        }
    }
    __syncthreads();

    // ---- Phase 4: dist = attn @ K  (values == keys in the reference) -----
    v8f zero = {};
    v8f dacc[4] = {zero, zero, zero, zero};
#pragma unroll
    for (int j = 0; j < 8; ++j) {
        int key0 = (w + 8 * j) * 32;
        v16h a = load_frag_a(attn_lds + (size_t)lc * LEN + key0, g);
#pragma unroll
        for (int nt = 0; nt < 4; ++nt) {
            v16h bb = load_frag_b(ktbase + (size_t)(nt * 16 + lc) * LEN + key0, g);
            dacc[nt] = wmma16(a, bb, dacc[nt]);
        }
    }
#pragma unroll
    for (int nt = 0; nt < 4; ++nt)
#pragma unroll
        for (int r = 0; r < 8; ++r)
            distp[(w * 16 + 8 * g + r) * 64 + nt * 16 + lc] = dacc[nt][r];
    __syncthreads();

    int tid = threadIdx.x;
#pragma unroll
    for (int u = 0; u < 4; ++u) {
        int idx = tid * 4 + u;     // 0..1023 over (row, dcol)
        int row = idx >> 6;
        int dc  = idx & 63;
        float s = 0.0f;
        for (int w2 = 0; w2 < 8; ++w2) s += distp[(w2 * 16 + row) * 64 + dc];
        __builtin_nontemporal_store(
            s, &out_dist[((size_t)b * LEN + q0 + row) * DMODEL + h * HD + dc]);
    }
}

// ---------------------------------------------------------------------------
// Launch
// ---------------------------------------------------------------------------
extern "C" void kernel_launch(void* const* d_in, const int* in_sizes, int n_in,
                              void* d_out, int out_size, void* d_ws, size_t ws_size,
                              hipStream_t stream) {
    const float* inp = (const float*)d_in[0];
    const float* Wq  = (const float*)d_in[1];
    const float* bq  = (const float*)d_in[2];
    const float* Wk  = (const float*)d_in[3];
    const float* bk  = (const float*)d_in[4];
    // d_in[5] = Wv, d_in[6] = bv : dead code in the reference (V is unused)

    // Workspace layout (f16), ~17 MiB total:
    char* ws = (char*)d_ws;
    _Float16* qk_h = (_Float16*)(ws);                          // B*L*F    = 4 MiB
    _Float16* wq_h = (_Float16*)(ws + (4u << 20));             // D*F      = 512 KiB
    _Float16* wk_h = (_Float16*)(ws + (4u << 20) + (512u << 10));
    _Float16* q_h  = (_Float16*)(ws + (5u << 20));             // (B,H,L,64)  4 MiB
    _Float16* kd_h = (_Float16*)(ws + (9u << 20));             // (B,H,L,64)  4 MiB
    _Float16* kt_h = (_Float16*)(ws + (13u << 20));            // (B,H,64,L)  4 MiB

    float* out_dist = (float*)d_out;                           // (B,L,512)
    float* out_attn = out_dist + (size_t)BATCH * LEN * DMODEL; // (B,L,L,H)

    int total = BATCH * LEN * FEAT;
    k_prepare<<<(total + 255) / 256, 256, 0, stream>>>(inp, Wq, Wk, qk_h, wq_h, wk_h);

    // 256 m-tiles * 32 n-tiles * 2 (Q,K) = 16384 waves; 8 waves/block
    k_proj<<<16384 / 8, 256, 0, stream>>>(qk_h, wq_h, wk_h, bq, bk, q_h, kd_h, kt_h);

    size_t smem = (size_t)16 * LEN * sizeof(_Float16)   // attn staging
                + (size_t)8 * 16 * 64 * sizeof(float)   // dist partials
                + (size_t)2 * 8 * 16 * sizeof(float);   // max/sum reductions
    k_attn<<<BATCH * NH * (LEN / 16), 256, smem, stream>>>(q_h, kd_h, kt_h,
                                                           out_dist, out_attn);
}
